// MS_SSA_Conv_39178691674143
// MI455X (gfx1250) — matrix operand order repaired and checked by reading, not compiled, not run on hardware
//
#include <hip/hip_runtime.h>
#include <hip/hip_bf16.h>

// MS-SSA block for MI455X (gfx1250, wave32, WMMA, LDS-staged GEMMs).
// All GEMM stages run through v_wmma_f32_16x16x32_bf16 (bf16 in, f32 acc).
// LDS slabs are stored TRANSPOSED (K-major per lane) so every WMMA fragment
// is a contiguous 32B per-lane read (ds_load_b128 pairs) instead of 16
// scalar ds_load_u16 + b16-mov packing ops.

typedef __bf16 bf16_t;
typedef __attribute__((ext_vector_type(16))) __bf16 v16bf;
typedef __attribute__((ext_vector_type(8)))  __bf16 v8bf;
typedef __attribute__((ext_vector_type(8)))  float  v8f;

#define T_   8
#define B_   16
#define C_   384
#define N_   1024      // H*W
#define HEADS_ 8
#define HD_  48        // C/HEADS
#define HDP_ 64        // padded head dim
#define S_   2048      // CHUNK*N
#define KSTR_ 48       // LDS row stride for 32-K slabs (96B: 32B-aligned k=0/16)
#define DSTR_ 80       // LDS row stride for 64-K slabs (160B: 32B-aligned k=0/16/32/48)

static __device__ __forceinline__ v8f wmma_bf16(const v16bf& a, const v16bf& b, const v8f& c) {
    return __builtin_amdgcn_wmma_f32_16x16x32_bf16(false, a, false, b, (short)0, c, false, false);
}

// swizzled K-position so the A-fragment pattern {kh..kh+7, 16+kh..16+kh+7}
// becomes one contiguous 16-element run per half-wave
static __device__ __forceinline__ int kswz(int s) {
    return (s & 7) | ((s & 16) >> 1) | ((s & 8) << 1);
}

// ---------------------------------------------------------------- prep ----
__global__ void prep_weights_kernel(const float* __restrict__ qw,
                                    const float* __restrict__ kw,
                                    const float* __restrict__ vw,
                                    const float* __restrict__ pw,
                                    bf16_t* __restrict__ out) {
    int tid = blockIdx.x * blockDim.x + threadIdx.x;
    if (tid >= 4 * C_ * C_) return;
    int g = tid / (C_ * C_), i = tid % (C_ * C_);
    const float* src = (g == 0) ? qw : ((g == 1) ? kw : ((g == 2) ? vw : pw));
    out[tid] = (bf16_t)src[i];
}

// bnp layout: for j in {q,k,v,proj}: inv[384], add[384]   (8*384 floats)
__global__ void prep_bn_kernel(const float* __restrict__ qbn,
                               const float* __restrict__ kbn,
                               const float* __restrict__ vbn,
                               const float* __restrict__ pbn,
                               const float* __restrict__ pb,
                               float* __restrict__ out) {
    int i = blockIdx.x * blockDim.x + threadIdx.x;
    if (i >= C_) return;
    const float* bns[4] = {qbn, kbn, vbn, pbn};
    #pragma unroll
    for (int j = 0; j < 4; ++j) {
        const float* bn = bns[j];
        float g  = bn[0 * C_ + i];
        float be = bn[1 * C_ + i];
        float mn = bn[2 * C_ + i];
        float vr = bn[3 * C_ + i];
        float inv = g * rsqrtf(vr + 1e-5f);
        float ad  = be - mn * inv;
        if (j == 3) ad += pb[i] * inv;      // fold proj bias through bn
        out[j * 768 + i]        = inv;
        out[j * 768 + C_ + i]   = ad;
    }
}

// ---------------------------------------------------------------- QKV ----
// grid: (128 tb, 3 m-groups, 16 n-tiles), block 256 (8 waves).
// Block tile: 128(M) x 64(N); K=384 in 12 steps. X slab staged transposed in
// LDS (relu+bf16 at staging), reused by all 8 waves and all 3 matrices.
__global__ __launch_bounds__(256) void qkv_kernel(
    const float*  __restrict__ x,
    const bf16_t* __restrict__ qw,
    const bf16_t* __restrict__ kw,
    const bf16_t* __restrict__ vw,
    const float*  __restrict__ bnp,
    bf16_t* __restrict__ q_ws,
    bf16_t* __restrict__ k_ws,
    float*  __restrict__ v_out) {
    __shared__ __align__(32) bf16_t xsT[64][KSTR_];   // [n][k] transposed slab

    const int tb   = blockIdx.x;
    const int t    = tb >> 4, b = tb & 15;
    const int tid  = threadIdx.x;
    const int lane = tid & 31;
    const int wv   = tid >> 5;
    const int m0   = (blockIdx.y * 8 + wv) * 16;
    const int n0   = blockIdx.z * 64;
    const int lan  = lane & 15;
    const bool hi  = lane >= 16;
    const int khalfA = hi ? 8 : 0;     // A-frag K split across half-waves
    const int kbaseB = hi ? 16 : 0;    // B-frag K split across half-waves

    // staging map: each thread owns 8 consecutive columns (n) of one slab row (k)
    const int lk = tid >> 3;           // slab k-row 0..31
    const int lc = (tid & 7) * 8;      // n col 0,8,...,56

    v8f acc[3][4];
    #pragma unroll
    for (int j = 0; j < 3; ++j)
        #pragma unroll
        for (int nf = 0; nf < 4; ++nf) { v8f z = {}; acc[j][nf] = z; }

    const float* xb = x + (size_t)tb * C_ * N_;

    for (int ko = 0; ko < C_; ko += 32) {
        // ---- cooperative stage: coalesced b128 loads -> relu/bf16 -> transpose scatter
        const float* src = xb + (size_t)(ko + lk) * N_ + n0 + lc;
        float4 f0 = *(const float4*)(src);
        float4 f1 = *(const float4*)(src + 4);
        xsT[lc + 0][lk] = (bf16_t)fmaxf(f0.x, 0.0f);
        xsT[lc + 1][lk] = (bf16_t)fmaxf(f0.y, 0.0f);
        xsT[lc + 2][lk] = (bf16_t)fmaxf(f0.z, 0.0f);
        xsT[lc + 3][lk] = (bf16_t)fmaxf(f0.w, 0.0f);
        xsT[lc + 4][lk] = (bf16_t)fmaxf(f1.x, 0.0f);
        xsT[lc + 5][lk] = (bf16_t)fmaxf(f1.y, 0.0f);
        xsT[lc + 6][lk] = (bf16_t)fmaxf(f1.z, 0.0f);
        xsT[lc + 7][lk] = (bf16_t)fmaxf(f1.w, 0.0f);
        __syncthreads();

        // ---- B fragments: contiguous 32B per-lane LDS reads
        v16bf bfr[4];
        #pragma unroll
        for (int nf = 0; nf < 4; ++nf)
            bfr[nf] = *(const v16bf*)&xsT[nf * 16 + lan][kbaseB];

        // ---- A fragments from global (weights, L2-resident) + WMMA
        #pragma unroll
        for (int j = 0; j < 3; ++j) {
            const bf16_t* wp = (j == 0) ? qw : ((j == 1) ? kw : vw);
            const bf16_t* ap = wp + (size_t)(m0 + lan) * C_ + ko + kbaseB;
            v16bf af;
            #pragma unroll
            for (int i = 0; i < 8; ++i) af[i]     = ap[i];        // K = kbase..kbase+7
            #pragma unroll
            for (int i = 0; i < 8; ++i) af[8 + i] = ap[8 + i];    // K = kbase+8..kbase+15
            #pragma unroll
            for (int nf = 0; nf < 4; ++nf)
                acc[j][nf] = wmma_bf16(af, bfr[nf], acc[j][nf]);
        }
        __syncthreads();
    }

    // epilogue: bn + relu, scatter to head layouts
    const int mbase = m0 + (hi ? 8 : 0);
    const int cch = t >> 1, chunk = t & 1;
    #pragma unroll
    for (int j = 0; j < 3; ++j) {
        const float* inv = bnp + j * 768;
        const float* add = inv + C_;
        #pragma unroll
        for (int r = 0; r < 8; ++r) {
            int m = mbase + r;
            float iv = inv[m], ad = add[m];
            int h = m / HD_, dd = m % HD_;
            #pragma unroll
            for (int nf = 0; nf < 4; ++nf) {
                int n = n0 + nf * 16 + lan;
                float val = fmaxf(acc[j][nf][r] * iv + ad, 0.0f);
                if (j == 2) {
                    // v -> second output segment, fp32, [T,B,8,1024,48]
                    v_out[(size_t)(((t * B_ + b) * HEADS_ + h) * N_ + n) * HD_ + dd] = val;
                } else {
                    size_t s   = (size_t)chunk * N_ + n;
                    size_t idx = ((size_t)((cch * B_ + b) * HEADS_ + h) * S_ + s) * HDP_ + dd;
                    (j == 0 ? q_ws : k_ws)[idx] = (bf16_t)val;
                }
            }
        }
    }
}

// ------------------------------------------------------------- attn = K^T V
// grid 512 (nc,b,h), block 128 (4 waves). Block computes full 64x64 attn tile,
// K = S = 2048. K slab stored [d][s] so A-fragments are contiguous;
// V slab stored [e][s] so B-fragments are contiguous. Pads zeroed at staging.
__global__ __launch_bounds__(128) void attn1_kernel(
    const bf16_t* __restrict__ k_ws,
    const float*  __restrict__ v_out,
    bf16_t* __restrict__ attn_ws) {
    __shared__ __align__(32) bf16_t ksT[64][KSTR_];   // [d][s]
    __shared__ __align__(32) bf16_t vsT[64][KSTR_];   // [e][s]

    const int cbh  = blockIdx.x;
    const int c    = cbh >> 7, b = (cbh >> 3) & 15, h = cbh & 7;
    const int tid  = threadIdx.x;
    const int lane = tid & 31;
    const int wv   = tid >> 5;
    const int lan  = lane & 15;
    const bool hi  = lane >= 16;
    const int khalfA = hi ? 8 : 0;
    const int kbaseB = hi ? 16 : 0;
    const int d = wv * 16 + lan;       // A-matrix row = head-dim index (padded)

    // staging map: 4 threads per s-row, 16 head-dim cols each
    const int lsr = tid >> 2;          // slab s-row 0..31
    const int lc  = (tid & 3) * 16;    // d/e col 0,16,32,48
    const int spK = lsr;               // s position for K slab

    v8f acc[4];
    #pragma unroll
    for (int ef = 0; ef < 4; ++ef) { v8f z = {}; acc[ef] = z; }

    const bf16_t* kb = k_ws + (size_t)cbh * S_ * HDP_;

    for (int so = 0; so < S_; so += 32) {
        // ---- stage K slab (zero d-pad rows), transpose scatter
        if (lc < HD_) {
            v16bf kv = *(const v16bf*)(kb + (size_t)(so + lsr) * HDP_ + lc);
            #pragma unroll
            for (int i = 0; i < 16; ++i) ksT[lc + i][spK] = kv[i];
        } else {
            #pragma unroll
            for (int i = 0; i < 16; ++i) ksT[lc + i][spK] = (bf16_t)0.0f;
        }
        // ---- stage V slab (fp32 -> bf16, zero e-pad rows), transpose scatter
        {
            int s   = so + lsr;
            int tt  = c * 2 + (s >> 10);
            int pix = s & 1023;
            if (lc < HD_) {
                const float* srcv =
                    v_out + (size_t)(((tt * B_ + b) * HEADS_ + h) * N_ + pix) * HD_ + lc;
                #pragma unroll
                for (int i = 0; i < 16; ++i) vsT[lc + i][lsr] = (bf16_t)srcv[i];
            } else {
                #pragma unroll
                for (int i = 0; i < 16; ++i) vsT[lc + i][lsr] = (bf16_t)0.0f;
            }
        }
        __syncthreads();

        // A = K^T (d x s): one contiguous 32B read (s = 16h + i, as for B)
        v16bf af = *(const v16bf*)&ksT[d][khalfA * 2];
        // B = V (s x e): contiguous 32B reads
        v16bf bfr[4];
        #pragma unroll
        for (int ef = 0; ef < 4; ++ef)
            bfr[ef] = *(const v16bf*)&vsT[ef * 16 + lan][kbaseB];
        #pragma unroll
        for (int ef = 0; ef < 4; ++ef)
            acc[ef] = wmma_bf16(af, bfr[ef], acc[ef]);
        __syncthreads();
    }

    const float scale = 1.0f / 32.0f;   // 1/H
    bf16_t* ab = attn_ws + (size_t)cbh * HDP_ * HDP_;
    const int mb = wv * 16 + (hi ? 8 : 0);
    #pragma unroll
    for (int r = 0; r < 8; ++r)
        #pragma unroll
        for (int ef = 0; ef < 4; ++ef)
            ab[(size_t)(mb + r) * HDP_ + ef * 16 + lan] = (bf16_t)(acc[ef][r] * scale);
}

// --------------------------------------------------------- out = Q * attn
// grid (512, 16), block 256 (8 waves). attn tile staged once, transposed
// [e][d] in LDS so B-fragments are contiguous; reused by all 8 waves.
__global__ __launch_bounds__(256) void attn2_kernel(
    const bf16_t* __restrict__ q_ws,
    const bf16_t* __restrict__ attn_ws,
    bf16_t* __restrict__ out_mid) {
    __shared__ __align__(32) bf16_t asT[64][DSTR_];   // [e][d]

    const int cbh  = blockIdx.x;
    const int c    = cbh >> 7, b = (cbh >> 3) & 15, h = cbh & 7;
    const int tid  = threadIdx.x;
    const int lane = tid & 31;
    const int wv   = tid >> 5;
    const int lan  = lane & 15;
    const bool hi  = lane >= 16;
    const int khalfA = hi ? 8 : 0;
    const int kbaseB = hi ? 16 : 0;
    const int s0 = (blockIdx.y * 8 + wv) * 16;

    const bf16_t* qb = q_ws    + (size_t)cbh * S_ * HDP_;
    const bf16_t* ab = attn_ws + (size_t)cbh * HDP_ * HDP_;

    // stage attn tile transposed: 256 threads, 4 per d-row, 16 e-cols each
    {
        const int ld = tid >> 2;            // d row 0..63
        const int le = (tid & 3) * 16;      // e col 0,16,32,48
        v16bf av = *(const v16bf*)(ab + (size_t)ld * HDP_ + le);
        #pragma unroll
        for (int i = 0; i < 16; ++i) asT[le + i][ld] = av[i];
    }
    __syncthreads();

    v8f acc[4];
    #pragma unroll
    for (int ef = 0; ef < 4; ++ef) { v8f z = {}; acc[ef] = z; }

    #pragma unroll
    for (int ko = 0; ko < HDP_; ko += 32) {
        v16bf af;                      // A = Q (s x d), contiguous d with pad zeroing
        const bf16_t* ap = qb + (size_t)(s0 + lan) * HDP_ + ko + kbaseB;
        #pragma unroll
        for (int i = 0; i < 8; ++i) {
            int dd = ko + kbaseB + i;
            af[i] = (dd < HD_) ? ap[i] : (bf16_t)0.0f;
        }
        #pragma unroll
        for (int i = 0; i < 8; ++i) {
            int dd = ko + kbaseB + 8 + i;
            af[8 + i] = (dd < HD_) ? ap[8 + i] : (bf16_t)0.0f;
        }
        v16bf bfr[4];                  // B = attn (d x e): contiguous 32B LDS reads
        #pragma unroll
        for (int ef = 0; ef < 4; ++ef) {
            bfr[ef] = *(const v16bf*)&asT[ef * 16 + lan][ko + kbaseB];
            acc[ef] = wmma_bf16(af, bfr[ef], acc[ef]);
        }
    }

    // relu + merge heads back to channels: out_mid[t,b, h*48+e, pix] (bf16)
    const int mb = s0 + (hi ? 8 : 0);
    #pragma unroll
    for (int r = 0; r < 8; ++r) {
        int s = mb + r;
        int chunk = s >> 10, pix = s & 1023;
        int t = c * 2 + chunk;
        #pragma unroll
        for (int ef = 0; ef < 4; ++ef) {
            int e = ef * 16 + lan;
            if (e < HD_) {
                float val = fmaxf(acc[ef][r], 0.0f);
                int o = h * HD_ + e;
                out_mid[((size_t)(t * B_ + b) * C_ + o) * N_ + pix] = (bf16_t)val;
            }
        }
    }
}

// ---------------------------------------------------------------- proj ----
// grid (128, 3, 16), block 256. y = bn(W_p * out_mid + b_p) + identity.
// out_mid slab (already bf16 + relu'd) staged transposed through LDS.
__global__ __launch_bounds__(256) void proj_kernel(
    const bf16_t* __restrict__ out_mid,
    const bf16_t* __restrict__ pw,
    const float*  __restrict__ bnp,   // pinv[384], padd[384] (bias folded)
    const float*  __restrict__ x,
    float* __restrict__ y_out) {
    __shared__ __align__(32) bf16_t xsT[64][KSTR_];   // [n][k]

    const int tb   = blockIdx.x;
    const int tid  = threadIdx.x;
    const int lane = tid & 31;
    const int wv   = tid >> 5;
    const int m0   = (blockIdx.y * 8 + wv) * 16;
    const int n0   = blockIdx.z * 64;
    const int lan  = lane & 15;
    const bool hi  = lane >= 16;
    const int khalfA = hi ? 8 : 0;
    const int kbaseB = hi ? 16 : 0;

    const int lk = tid >> 3;
    const int lc = (tid & 7) * 8;

    v8f acc[4];
    #pragma unroll
    for (int nf = 0; nf < 4; ++nf) { v8f z = {}; acc[nf] = z; }

    const bf16_t* xb = out_mid + (size_t)tb * C_ * N_;

    for (int ko = 0; ko < C_; ko += 32) {
        v8bf pk = *(const v8bf*)(xb + (size_t)(ko + lk) * N_ + n0 + lc);
        #pragma unroll
        for (int i = 0; i < 8; ++i) xsT[lc + i][lk] = pk[i];
        __syncthreads();

        v16bf bfr[4];
        #pragma unroll
        for (int nf = 0; nf < 4; ++nf)
            bfr[nf] = *(const v16bf*)&xsT[nf * 16 + lan][kbaseB];

        const bf16_t* ap = pw + (size_t)(m0 + lan) * C_ + ko + kbaseB;
        v16bf af;
        #pragma unroll
        for (int i = 0; i < 8; ++i) af[i]     = ap[i];
        #pragma unroll
        for (int i = 0; i < 8; ++i) af[8 + i] = ap[8 + i];
        #pragma unroll
        for (int nf = 0; nf < 4; ++nf)
            acc[nf] = wmma_bf16(af, bfr[nf], acc[nf]);
        __syncthreads();
    }

    const int mbase = m0 + (hi ? 8 : 0);
    const float* inv = bnp;
    const float* add = bnp + C_;
    const float* idb = x + (size_t)tb * C_ * N_;
    float* yb = y_out + (size_t)tb * C_ * N_;
    #pragma unroll
    for (int r = 0; r < 8; ++r) {
        int m = mbase + r;
        float iv = inv[m], ad = add[m];
        #pragma unroll
        for (int nf = 0; nf < 4; ++nf) {
            int n = n0 + nf * 16 + lan;
            size_t idx = (size_t)m * N_ + n;
            yb[idx] = acc[nf][r] * iv + ad + idb[idx];
        }
    }
}

// ------------------------------------------------------------- launcher ----
extern "C" void kernel_launch(void* const* d_in, const int* in_sizes, int n_in,
                              void* d_out, int out_size, void* d_ws, size_t ws_size,
                              hipStream_t stream) {
    const float* x   = (const float*)d_in[0];
    const float* qwf = (const float*)d_in[1];
    const float* qbn = (const float*)d_in[2];
    const float* kwf = (const float*)d_in[3];
    const float* kbn = (const float*)d_in[4];
    const float* vwf = (const float*)d_in[5];
    const float* vbn = (const float*)d_in[6];
    const float* pwf = (const float*)d_in[7];
    const float* pb  = (const float*)d_in[8];
    const float* pbn = (const float*)d_in[9];

    const size_t Y_ELEMS = (size_t)T_ * B_ * C_ * N_;         // 50,331,648
    float* y_out = (float*)d_out;
    float* v_out = y_out + Y_ELEMS;

    char* ws = (char*)d_ws;
    bf16_t* w_bf = (bf16_t*)ws;                               // 4 * 384*384 bf16
    float*  bnp  = (float*)(ws + 4ull * C_ * C_ * 2);         // 8 * 384 f32
    char*   big  = ws + 4ull * C_ * C_ * 2 + 8ull * C_ * 4;   // = 1,191,936 (32B aligned)
    bf16_t* q_ws    = (bf16_t*)big;                           // [512][2048][64] bf16
    bf16_t* k_ws    = q_ws + (size_t)512 * S_ * HDP_;
    bf16_t* attn_ws = k_ws + (size_t)512 * S_ * HDP_;         // [512][64][64] bf16
    bf16_t* out_mid = attn_ws + (size_t)512 * HDP_ * HDP_;    // [T,B,C,N] bf16

    prep_weights_kernel<<<(4 * C_ * C_ + 255) / 256, 256, 0, stream>>>(qwf, kwf, vwf, pwf, w_bf);
    prep_bn_kernel<<<2, 256, 0, stream>>>(qbn, kbn, vbn, pbn, pb, bnp);

    dim3 g1(T_ * B_, 3, 16);
    qkv_kernel<<<g1, 256, 0, stream>>>(x, w_bf, w_bf + C_ * C_, w_bf + 2 * C_ * C_,
                                       bnp, q_ws, k_ws, v_out);

    attn1_kernel<<<512, 128, 0, stream>>>(k_ws, v_out, attn_ws);

    dim3 g2(512, 16);
    attn2_kernel<<<g2, 256, 0, stream>>>(q_ws, attn_ws, out_mid);

    dim3 g3(T_ * B_, 3, 16);
    proj_kernel<<<g3, 256, 0, stream>>>(out_mid, w_bf + 3 * C_ * C_, bnp + 3 * 768, x, y_out);
}
